// DecoderTreeLSTM_35862976921781
// MI455X (gfx1250) — matrix operand order repaired and checked by reading, not compiled, not run on hardware
//
#include <hip/hip_runtime.h>
#include <hip/hip_bf16.h>

// ---------------------------------------------------------------------------
// Tree-LSTM decoder on gfx1250 (MI455X), wave32 + V_WMMA_F32_16X16X32_F16.
// Levels sequential; per level: gather -> fused GEMM+LSTM cell -> out GEMM+argmax.
// Weights pre-transposed to column-major fp16 so WMMA B-fragments are
// contiguous 32B loads per lane; A-fragments are two contiguous 16B loads.
// ---------------------------------------------------------------------------

typedef __attribute__((ext_vector_type(16))) _Float16 v16h;
typedef __attribute__((ext_vector_type(8)))  _Float16 v8h;
typedef __attribute__((ext_vector_type(8)))  float    v8f;

#define LEVEL_W 512
#define DEPTH   16
#define NNODE   (LEVEL_W*DEPTH)
#define DIN     512
#define HDIM    512
#define EDIM    200
#define NCLS    150
#define KTOT    1248            // 512 feat + 200 embed + 24 pad + 512 h
#define KSTEPS  (KTOT/32)       // 39
#define NG      2048            // 4*HDIM gate width
#define NCPAD   160             // classes padded to 10 WMMA tiles

union V16 { v16h v; v8h h[2]; };
union V8F { v8f v; float f[8]; };

__device__ __forceinline__ float sigf(float x) { return 1.0f / (1.0f + __expf(-x)); }

// --- one-time: W combined, transposed to column-major fp16: Wt[n][k] --------
__global__ void conv_w_kernel(const float* __restrict__ Wx,
                              const float* __restrict__ Wh,
                              _Float16* __restrict__ Wt) {
    int n = blockIdx.x;                       // 0..2047
    for (int k = threadIdx.x; k < KTOT; k += blockDim.x) {
        float v;
        if (k < DIN + EDIM)      v = Wx[(size_t)k * NG + n];
        else if (k < 736)        v = 0.0f;                     // K padding
        else                     v = Wh[(size_t)(k - 736) * NG + n];
        Wt[(size_t)n * KTOT + k] = (_Float16)v;
    }
}

// --- one-time: out_W transposed column-major fp16, padded bias --------------
__global__ void conv_outw_kernel(const float* __restrict__ outW,
                                 const float* __restrict__ outb,
                                 _Float16* __restrict__ oWt,
                                 float* __restrict__ obp) {
    int n = blockIdx.x;                       // 0..159
    for (int k = threadIdx.x; k < HDIM; k += blockDim.x)
        oWt[(size_t)n * HDIM + k] = (_Float16)(n < NCLS ? outW[(size_t)k * NCLS + n] : 0.0f);
    if (threadIdx.x == 0) obp[n] = (n < NCLS) ? outb[n] : -1e30f;
}

// --- per level: build fp16 input rows [feat | embed(parent label) | 0 | h_par]
__global__ void gather_kernel(const float* __restrict__ feat,
                              const float* __restrict__ emb,
                              const int*   __restrict__ parent,
                              const int*   __restrict__ cm_all,
                              const _Float16* __restrict__ h_prev,
                              _Float16* __restrict__ Xg, int off) {
    int r = blockIdx.x;                       // 0..511
    int node = off + r;
    int p = parent[node];
    bool root = (p < 0);
    int eidx = root ? 0 : (cm_all[p] + 1);
    int pl = root ? 0 : (p - (off - LEVEL_W));
    _Float16* xr = Xg + (size_t)r * KTOT;
    const float* fr = feat + (size_t)node * DIN;
    for (int c = threadIdx.x; c < DIN; c += blockDim.x)
        xr[c] = (_Float16)fr[c];
    for (int c = threadIdx.x; c < 224; c += blockDim.x)
        xr[512 + c] = (_Float16)(c < EDIM ? emb[(size_t)eidx * EDIM + c] : 0.0f);
    for (int c = threadIdx.x; c < HDIM; c += blockDim.x)
        xr[736 + c] = root ? (_Float16)0.0f : h_prev[(size_t)pl * HDIM + c];
}

// --- per level: G = [X|h] @ [Wx;Wh], fused LSTM elementwise -----------------
// grid (32 M-tiles, 8 j-blocks), 256 threads = 8 waves.
// wave w: gate = w>>1, half = w&1 -> two 16x16 tiles of G, K-loop 39 WMMAs x2.
__global__ __launch_bounds__(256) void cell_kernel(
    const _Float16* __restrict__ Xg, const _Float16* __restrict__ Wt,
    const float* __restrict__ b, const float* __restrict__ drop,
    const int* __restrict__ parent, const float* __restrict__ c_prev,
    float* __restrict__ c_cur, _Float16* __restrict__ h_half, int off) {
    __shared__ float G[16][260];              // 16 rows x (4 gates * 64 j), padded
    const int mt = blockIdx.x, jb = blockIdx.y;
    const int tid = threadIdx.x;
    const int wave = tid >> 5, lane = tid & 31;
    const int gate = wave >> 1, hw = wave & 1;
    const int laneN = lane & 15, hi = lane >> 4;
    const int colbase = gate * 64 + hw * 32;               // col in LDS tile
    const int N0 = gate * HDIM + jb * 64 + hw * 32;        // col in G (2048)

    const _Float16* arow = Xg + (size_t)(mt * 16 + laneN) * KTOT;
    const _Float16* b0p  = Wt + (size_t)(N0 + laneN) * KTOT;
    const _Float16* b1p  = Wt + (size_t)(N0 + 16 + laneN) * KTOT;

    v8f acc0 = {}; v8f acc1 = {};
    for (int kt = 0; kt < KSTEPS; ++kt) {
        const int kb = kt * 32;
        V16 a, bb0, bb1;
        // A fragment, contiguous k order chosen here (same as B): lane row 16 K-values at hi*16
        const v8h* pa = (const v8h*)(arow + kb + hi * 16);
        a.h[0] = pa[0]; a.h[1] = pa[1];
        // B fragment, same contiguous k order as A: lane column 16 K-values at hi*16
        const v8h* pb0 = (const v8h*)(b0p + kb + hi * 16);
        bb0.h[0] = pb0[0]; bb0.h[1] = pb0[1];
        const v8h* pb1 = (const v8h*)(b1p + kb + hi * 16);
        bb1.h[0] = pb1[0]; bb1.h[1] = pb1[1];
        __builtin_prefetch(b0p + kb + 128, 0, 1);          // global_prefetch_b8
        acc0 = __builtin_amdgcn_wmma_f32_16x16x32_f16(false, a.v, false, bb0.v,
                                                      (short)0, acc0, false, false);
        acc1 = __builtin_amdgcn_wmma_f32_16x16x32_f16(false, a.v, false, bb1.v,
                                                      (short)0, acc1, false, false);
    }
    // stage C/D (lane N = lane%16, vgpr v -> row v + hi*8) to LDS
    V8F u0, u1; u0.v = acc0; u1.v = acc1;
#pragma unroll
    for (int v = 0; v < 8; ++v) {
        G[v + hi * 8][colbase + laneN]      = u0.f[v];
        G[v + hi * 8][colbase + 16 + laneN] = u1.f[v];
    }
    __syncthreads();
    // elementwise LSTM: 16 rows x 64 j units, 4 per thread
#pragma unroll
    for (int q = 0; q < 4; ++q) {
        const int flat = tid * 4 + q;
        const int r = flat >> 6, j = flat & 63;
        const int jg = jb * 64 + j;
        const float gi = G[r][j]           + b[jg];
        const float gf = G[r][64 + j]      + b[HDIM + jg];
        const float gg = G[r][128 + j]     + b[2 * HDIM + jg];
        const float go = G[r][192 + j]     + b[3 * HDIM + jg];
        const int rl = mt * 16 + r;
        const int node = off + rl;
        const int p = parent[node];
        const float cp = (p < 0) ? 0.0f
                                 : c_prev[(size_t)(p - (off - LEVEL_W)) * HDIM + jg];
        const float c2 = sigf(gf) * cp + sigf(gi) * tanhf(gg);
        const float h2 = sigf(go) * tanhf(c2) * drop[jg];
        c_cur[(size_t)rl * HDIM + jg] = c2;
        h_half[(size_t)rl * HDIM + jg] = (_Float16)h2;
    }
}

// --- per level: dist = h @ out_W + b, argmax ---------------------------------
// grid 32 M-tiles, 320 threads = 10 waves (one 16x16 N-tile each).
__global__ __launch_bounds__(320) void out_kernel(
    const _Float16* __restrict__ h_half, const _Float16* __restrict__ oWt,
    const float* __restrict__ obp, float* __restrict__ d_all,
    int* __restrict__ cm_all, int off) {
    __shared__ float D[16][NCPAD + 4];
    const int mt = blockIdx.x;
    const int tid = threadIdx.x;
    const int wave = tid >> 5, lane = tid & 31;
    const int laneN = lane & 15, hi = lane >> 4;
    const int N0 = wave * 16;
    const _Float16* arow = h_half + (size_t)(mt * 16 + laneN) * HDIM;
    const _Float16* bp   = oWt + (size_t)(N0 + laneN) * HDIM;
    v8f acc = {};
    for (int kt = 0; kt < HDIM / 32; ++kt) {
        const int kb = kt * 32;
        V16 a, bb;
        const v8h* pa = (const v8h*)(arow + kb + hi * 16);
        a.h[0] = pa[0]; a.h[1] = pa[1];
        const v8h* pb = (const v8h*)(bp + kb + hi * 16);
        bb.h[0] = pb[0]; bb.h[1] = pb[1];
        acc = __builtin_amdgcn_wmma_f32_16x16x32_f16(false, a.v, false, bb.v,
                                                     (short)0, acc, false, false);
    }
    V8F u; u.v = acc;
    const int col = N0 + laneN;
    const float bias = obp[col];
#pragma unroll
    for (int v = 0; v < 8; ++v) {
        const int r = v + hi * 8;
        const float val = u.f[v] + bias;
        D[r][col] = val;
        if (col < NCLS) {
            const int node = off + mt * 16 + r;
            d_all[(size_t)node * NCLS + col] = val;
        }
    }
    __syncthreads();
    if (tid < 16) {
        const int r = tid;
        float best = D[r][0]; int bi = 0;
        for (int c = 1; c < NCLS; ++c) {       // first-occurrence argmax
            const float x = D[r][c];
            if (x > best) { best = x; bi = c; }
        }
        cm_all[off + mt * 16 + r] = bi;
    }
}

// --- final: apply `order` permutation, emit dists + commitments -------------
__global__ void reorder_kernel(const float* __restrict__ d_all,
                               const int* __restrict__ cm_all,
                               const int* __restrict__ order,
                               float* __restrict__ out, int nout) {
    const int i = blockIdx.x;
    const int src = order[i];
    if (threadIdx.x < NCLS)
        out[(size_t)i * NCLS + threadIdx.x] = d_all[(size_t)src * NCLS + threadIdx.x];
    if (threadIdx.x == 0)
        out[(size_t)nout * NCLS + i] = (float)cm_all[src];
}

extern "C" void kernel_launch(void* const* d_in, const int* in_sizes, int n_in,
                              void* d_out, int out_size, void* d_ws, size_t ws_size,
                              hipStream_t stream) {
    (void)in_sizes; (void)n_in; (void)out_size; (void)ws_size;
    const float* feat   = (const float*)d_in[0];
    const float* emb    = (const float*)d_in[1];
    const float* Wx     = (const float*)d_in[2];
    const float* Wh     = (const float*)d_in[3];
    const float* b      = (const float*)d_in[4];
    const float* outW   = (const float*)d_in[5];
    const float* outb   = (const float*)d_in[6];
    const float* drop   = (const float*)d_in[7];
    const int*   parent = (const int*)d_in[8];
    const int*   order  = (const int*)d_in[9];

    char* ws = (char*)d_ws;
    size_t o = 0;
    _Float16* Wt  = (_Float16*)(ws + o); o += (size_t)NG * KTOT * 2;        // 5.11 MB
    _Float16* oWt = (_Float16*)(ws + o); o += (size_t)NCPAD * HDIM * 2;     // 160 KB
    float*    obp = (float*)(ws + o);    o += 1024;
    _Float16* Xg  = (_Float16*)(ws + o); o += (size_t)LEVEL_W * KTOT * 2;   // 1.28 MB
    _Float16* hbuf[2];
    hbuf[0] = (_Float16*)(ws + o); o += (size_t)LEVEL_W * HDIM * 2;
    hbuf[1] = (_Float16*)(ws + o); o += (size_t)LEVEL_W * HDIM * 2;
    float* cbuf[2];
    cbuf[0] = (float*)(ws + o); o += (size_t)LEVEL_W * HDIM * 4;
    cbuf[1] = (float*)(ws + o); o += (size_t)LEVEL_W * HDIM * 4;
    int*   cm   = (int*)(ws + o);   o += (size_t)NNODE * 4;
    float* dall = (float*)(ws + o); o += (size_t)NNODE * NCLS * 4;          // ~14.6 MB total

    conv_w_kernel<<<NG, 256, 0, stream>>>(Wx, Wh, Wt);
    conv_outw_kernel<<<NCPAD, 256, 0, stream>>>(outW, outb, oWt, obp);

    for (int l = 0; l < DEPTH; ++l) {
        const int off = l * LEVEL_W;
        _Float16* hin  = hbuf[(l + 1) & 1];
        _Float16* hout = hbuf[l & 1];
        float*    cin  = cbuf[(l + 1) & 1];
        float*    cout = cbuf[l & 1];
        gather_kernel<<<LEVEL_W, 128, 0, stream>>>(feat, emb, parent, cm, hin, Xg, off);
        cell_kernel<<<dim3(32, 8), 256, 0, stream>>>(Xg, Wt, b, drop, parent,
                                                     cin, cout, hout, off);
        out_kernel<<<32, 320, 0, stream>>>(hout, oWt, obp, dall, cm, off);
    }

    const int nout = NNODE - LEVEL_W;
    reorder_kernel<<<nout, 160, 0, stream>>>(dall, cm, order, (float*)d_out, nout);
}
